// MultiHeadAttention_83288005804903
// MI455X (gfx1250) — matrix operand order repaired and checked by reading, not compiled, not run on hardware
//
#include <hip/hip_runtime.h>
#include <hip/hip_bf16.h>

typedef __bf16 bf16_t;
typedef bf16_t v16bf __attribute__((ext_vector_type(16)));
typedef bf16_t v8bf  __attribute__((ext_vector_type(8)));
typedef float  v8f   __attribute__((ext_vector_type(8)));

#define B_SZ  4
#define S_SZ  2048
#define D_SZ  1024
#define H_SZ  16
#define DH_SZ 64

// fp32 -> bf16 round-to-nearest-even
static __device__ __forceinline__ bf16_t f2bf(float f) {
  unsigned u = __builtin_bit_cast(unsigned, f);
  u += 0x7fffu + ((u >> 16) & 1u);
  unsigned short h = (unsigned short)(u >> 16);
  return __builtin_bit_cast(bf16_t, h);
}

static __device__ __forceinline__ v16bf cat8(v8bf lo, v8bf hi) {
  v16bf r;
#pragma unroll
  for (int i = 0; i < 8; ++i) { r[i] = lo[i]; r[i + 8] = hi[i]; }
  return r;
}

#define WMMA_BF16(A, Bm, C) \
  __builtin_amdgcn_wmma_f32_16x16x32_bf16(false, (A), false, (Bm), (short)0, (C), false, false)

// Async global->LDS copy, 16B per active lane.  LDS address = low 32 bits of
// the generic pointer (ISA 10.2: LDS aperture addr = addr[31:0]).  Tracked by
// ASYNCcnt; completion enforced with s_wait_asynccnt below.
static __device__ __forceinline__ void async_b128(void* lds_p, const void* g_p) {
  unsigned l = (unsigned)(unsigned long long)(lds_p);
  unsigned long long g = (unsigned long long)(g_p);
  asm volatile("global_load_async_to_lds_b128 %0, %1, off"
               :: "v"(l), "v"(g) : "memory");
}
static __device__ __forceinline__ void wait_async0() {
  asm volatile("s_wait_asynccnt 0x0" ::: "memory");
}
static __device__ __forceinline__ void wait_async3() {
  asm volatile("s_wait_asynccnt 0x3" ::: "memory");
}

// ---------------------------------------------------------------------------
// Pass 0a: x fp32 -> bf16 row-major.  Vectorized: 2x float4 in, 1x 16B out.
// ---------------------------------------------------------------------------
__global__ __launch_bounds__(256) void cvt_x_kernel(const float* __restrict__ x,
                                                    bf16_t* __restrict__ xb) {
  int idx = (blockIdx.x * 256 + threadIdx.x) * 8;
  float4 a = *(const float4*)(x + idx);
  float4 b = *(const float4*)(x + idx + 4);
  v8bf o;
  o[0] = f2bf(a.x); o[1] = f2bf(a.y); o[2] = f2bf(a.z); o[3] = f2bf(a.w);
  o[4] = f2bf(b.x); o[5] = f2bf(b.y); o[6] = f2bf(b.z); o[7] = f2bf(b.w);
  *(v8bf*)(xb + idx) = o;
}

// ---------------------------------------------------------------------------
// Pass 0b: W fp32 [K][N] -> bf16 transposed [N][K] (LDS-tiled so both global
// sides stay coalesced).  blockIdx.z selects {Wq,Wk,Wv}.
// ---------------------------------------------------------------------------
__global__ __launch_bounds__(256) void cvt_w_kernel(
    const float* __restrict__ Wq, const float* __restrict__ Wk,
    const float* __restrict__ Wv, bf16_t* __restrict__ Wt) {
  __shared__ bf16_t tile[32][33];
  const int w = blockIdx.z;
  const float* W = (w == 0) ? Wq : (w == 1) ? Wk : Wv;
  bf16_t* dst = Wt + (size_t)w * D_SZ * D_SZ;
  const int n0 = blockIdx.x * 32, k0 = blockIdx.y * 32;
  const int c = threadIdx.x & 31, r8 = threadIdx.x >> 5;
#pragma unroll
  for (int i = 0; i < 4; ++i) {
    int row = i * 8 + r8;
    tile[row][c] = f2bf(W[(size_t)(k0 + row) * D_SZ + n0 + c]);
  }
  __syncthreads();
#pragma unroll
  for (int i = 0; i < 4; ++i) {
    int row = i * 8 + r8;
    dst[(size_t)(n0 + row) * D_SZ + k0 + c] = tile[c][row];  // Wt[n][k] = W[k][n]
  }
}

// ---------------------------------------------------------------------------
// Pass 1: fused QKV projection on pre-converted bf16 operands.
// 256 threads / 8 waves; 128x64 block tile; K-step 32; double-buffered LDS
// filled by GLOBAL_LOAD_ASYNC_TO_LDS_B128 (3 async instrs per wave per tile).
// Q,K out [B,H,S,Dh] bf16; V out transposed [B,H,Dh,S] bf16.
// ---------------------------------------------------------------------------
__global__ __launch_bounds__(256) void qkv_gemm_kernel(
    const bf16_t* __restrict__ xb, const bf16_t* __restrict__ Wt,
    const float* __restrict__ bq, const float* __restrict__ bk, const float* __restrict__ bv,
    bf16_t* __restrict__ Qo, bf16_t* __restrict__ Ko, bf16_t* __restrict__ Vt)
{
  __shared__ bf16_t As[2][128][40];  // padded stride 80B: conflict-free frag reads
  __shared__ bf16_t Bs[2][64][40];   // Wt rows: [n][k]

  const int w = blockIdx.z;
  const bf16_t* Ww   = Wt + (size_t)w * D_SZ * D_SZ;
  const float*  bias = (w == 0) ? bq : (w == 1) ? bk : bv;

  const int tid   = threadIdx.x;
  const int lane  = tid & 31;
  const int wave  = tid >> 5;
  const int waveM = wave & 3;
  const int waveN = wave >> 2;
  const int lh = lane >> 4;
  const int ll = lane & 15;

  const int blockM = blockIdx.y * 128;
  const int blockN = blockIdx.x * 64;

  v8f acc[2][2];
#pragma unroll
  for (int i = 0; i < 2; ++i)
#pragma unroll
    for (int j = 0; j < 2; ++j)
#pragma unroll
      for (int r = 0; r < 8; ++r) acc[i][j][r] = 0.0f;

  // issue one K-tile's async copies: A 128x32 (512 16B chunks), B 64x32 (256)
  auto issue_tile = [&](int kk, int buf) {
#pragma unroll
    for (int i = 0; i < 2; ++i) {
      int c = i * 256 + tid;          // 0..511
      int row = c >> 2, ch = c & 3;
      async_b128(&As[buf][row][ch * 8],
                 xb + (size_t)(blockM + row) * D_SZ + kk + ch * 8);
    }
    {
      int row = tid >> 2, ch = tid & 3;
      async_b128(&Bs[buf][row][ch * 8],
                 Ww + (size_t)(blockN + row) * D_SZ + kk + ch * 8);
    }
  };

  const int NT = D_SZ / 32;
  issue_tile(0, 0);

  for (int t = 0; t < NT; ++t) {
    const int buf = t & 1;
    if (t + 1 < NT) {
      issue_tile((t + 1) * 32, buf ^ 1);  // prefetch next tile into other buffer
      wait_async3();                      // previous 3 ops (this tile) complete
    } else {
      wait_async0();
    }
    __syncthreads();

    v16bf af[2], bfg[2];
#pragma unroll
    for (int i = 0; i < 2; ++i) {
      int row = waveM * 32 + i * 16 + ll;
      v8bf lo = *(const v8bf*)&As[buf][row][lh * 8];
      v8bf hi = *(const v8bf*)&As[buf][row][16 + lh * 8];
      af[i] = cat8(lo, hi);
    }
#pragma unroll
    for (int j = 0; j < 2; ++j) {
      int colB = waveN * 32 + j * 16 + ll;
      v8bf lo = *(const v8bf*)&Bs[buf][colB][lh * 8];
      v8bf hi = *(const v8bf*)&Bs[buf][colB][16 + lh * 8];
      bfg[j] = cat8(lo, hi);
    }
#pragma unroll
    for (int i = 0; i < 2; ++i)
#pragma unroll
      for (int j = 0; j < 2; ++j)
        acc[i][j] = WMMA_BF16(af[i], bfg[j], acc[i][j]);
    __syncthreads();
  }

  // epilogue: bias add + scatter to head-major bf16 layouts
#pragma unroll
  for (int i = 0; i < 2; ++i) {
#pragma unroll
    for (int j = 0; j < 2; ++j) {
#pragma unroll
      for (int r = 0; r < 8; ++r) {
        int m = blockM + waveM * 32 + i * 16 + lh * 8 + r;  // C layout: M = r + 8*laneHi
        int n = blockN + waveN * 32 + j * 16 + ll;
        float v = acc[i][j][r] + bias[n];
        int bb = m >> 11;
        int ss = m & (S_SZ - 1);
        int hh = n >> 6;
        int dd = n & (DH_SZ - 1);
        if (w == 2) {
          Vt[((bb * H_SZ + hh) * DH_SZ + dd) * S_SZ + ss] = f2bf(v);
        } else {
          bf16_t* dst = (w == 0) ? Qo : Ko;
          dst[((bb * H_SZ + hh) * S_SZ + ss) * DH_SZ + dd] = f2bf(v);
        }
      }
    }
  }
}

// ---------------------------------------------------------------------------
// Pass 2: flash attention.  One wave per 16-query tile; key blocks of 32;
// QK^T and PV on WMMA; online softmax in fp32.
// ---------------------------------------------------------------------------
__global__ __launch_bounds__(32) void attn_kernel(
    const bf16_t* __restrict__ Q, const bf16_t* __restrict__ K,
    const bf16_t* __restrict__ Vt, float* __restrict__ out)
{
  __shared__ bf16_t Pl[16][40];  // P tile: C-layout -> A-layout reshape

  const int qt = blockIdx.x;
  const int h  = blockIdx.y;
  const int b  = blockIdx.z;
  const int lane = threadIdx.x;
  const int lh = lane >> 4;
  const int ll = lane & 15;

  const bf16_t* Qbh = Q  + (size_t)((b * H_SZ + h) * S_SZ) * DH_SZ;
  const bf16_t* Kbh = K  + (size_t)((b * H_SZ + h) * S_SZ) * DH_SZ;
  const bf16_t* Vbh = Vt + (size_t)((b * H_SZ + h) * DH_SZ) * S_SZ;

  const float scale = 0.03125f;  // 1/sqrt(D_MODEL)

  const bf16_t* qp = Qbh + (qt * 16 + ll) * DH_SZ;
  v16bf aq0 = cat8(*(const v8bf*)(qp + lh * 8),      *(const v8bf*)(qp + 16 + lh * 8));
  v16bf aq1 = cat8(*(const v8bf*)(qp + 32 + lh * 8), *(const v8bf*)(qp + 48 + lh * 8));

  v8f o[4];
  float mrow[8], lrow[8];
#pragma unroll
  for (int g = 0; g < 4; ++g)
#pragma unroll
    for (int r = 0; r < 8; ++r) o[g][r] = 0.0f;
#pragma unroll
  for (int r = 0; r < 8; ++r) { mrow[r] = -1e30f; lrow[r] = 0.0f; }

  for (int kb = 0; kb < S_SZ; kb += 32) {
    v8f c0, c1;
#pragma unroll
    for (int r = 0; r < 8; ++r) { c0[r] = 0.0f; c1[r] = 0.0f; }

    const bf16_t* kp0 = Kbh + (kb + ll) * DH_SZ;
    const bf16_t* kp1 = Kbh + (kb + 16 + ll) * DH_SZ;
    v16bf bk00 = cat8(*(const v8bf*)(kp0 + lh * 8),      *(const v8bf*)(kp0 + 16 + lh * 8));
    v16bf bk01 = cat8(*(const v8bf*)(kp0 + 32 + lh * 8), *(const v8bf*)(kp0 + 48 + lh * 8));
    v16bf bk10 = cat8(*(const v8bf*)(kp1 + lh * 8),      *(const v8bf*)(kp1 + 16 + lh * 8));
    v16bf bk11 = cat8(*(const v8bf*)(kp1 + 32 + lh * 8), *(const v8bf*)(kp1 + 48 + lh * 8));

    c0 = WMMA_BF16(aq0, bk00, c0);
    c0 = WMMA_BF16(aq1, bk01, c0);
    c1 = WMMA_BF16(aq0, bk10, c1);
    c1 = WMMA_BF16(aq1, bk11, c1);

#pragma unroll
    for (int r = 0; r < 8; ++r) {
      float s0 = c0[r] * scale;
      float s1 = c1[r] * scale;
      float mx = fmaxf(s0, s1);
#pragma unroll
      for (int off = 1; off < 16; off <<= 1)
        mx = fmaxf(mx, __shfl_xor(mx, off, 32));
      float mnew = fmaxf(mrow[r], mx);
      float fscl = __expf(mrow[r] - mnew);
      float p0 = __expf(s0 - mnew);
      float p1 = __expf(s1 - mnew);
      float ps = p0 + p1;
#pragma unroll
      for (int off = 1; off < 16; off <<= 1)
        ps += __shfl_xor(ps, off, 32);
      lrow[r] = lrow[r] * fscl + ps;
      mrow[r] = mnew;
#pragma unroll
      for (int g = 0; g < 4; ++g) o[g][r] *= fscl;
      Pl[lh * 8 + r][ll]      = f2bf(p0);
      Pl[lh * 8 + r][16 + ll] = f2bf(p1);
    }
    __syncthreads();

    v16bf ap = cat8(*(const v8bf*)&Pl[ll][lh * 8], *(const v8bf*)&Pl[ll][16 + lh * 8]);
#pragma unroll
    for (int g = 0; g < 4; ++g) {
      const bf16_t* vp = Vbh + (g * 16 + ll) * S_SZ + kb + lh * 8;
      v16bf bv = cat8(*(const v8bf*)vp, *(const v8bf*)(vp + 16));
      o[g] = WMMA_BF16(ap, bv, o[g]);
    }
    __syncthreads();
  }

#pragma unroll
  for (int r = 0; r < 8; ++r) {
    float inv = 1.0f / lrow[r];
    int q = qt * 16 + lh * 8 + r;
#pragma unroll
    for (int g = 0; g < 4; ++g) {
      int col = h * DH_SZ + g * 16 + ll;
      out[(b * S_SZ + q) * D_SZ + col] = o[g][r] * inv;
    }
  }
}

extern "C" void kernel_launch(void* const* d_in, const int* in_sizes, int n_in,
                              void* d_out, int out_size, void* d_ws, size_t ws_size,
                              hipStream_t stream) {
  (void)in_sizes; (void)n_in; (void)out_size; (void)ws_size;
  const float* x  = (const float*)d_in[0];
  const float* Wq = (const float*)d_in[1];
  const float* bq = (const float*)d_in[2];
  const float* Wk = (const float*)d_in[3];
  const float* bk = (const float*)d_in[4];
  const float* Wv = (const float*)d_in[5];
  const float* bv = (const float*)d_in[6];
  float* out = (float*)d_out;

  const size_t qkvElems = (size_t)B_SZ * H_SZ * S_SZ * DH_SZ;  // 8Mi
  const size_t xElems   = (size_t)B_SZ * S_SZ * D_SZ;          // 8Mi
  bf16_t* Qb  = (bf16_t*)d_ws;
  bf16_t* Kb  = Qb + qkvElems;
  bf16_t* Vt  = Kb + qkvElems;
  bf16_t* xbf = Vt + qkvElems;
  bf16_t* Wtb = xbf + xElems;     // 3 transposed weight matrices, bf16

  cvt_x_kernel<<<dim3((unsigned)(xElems / (256 * 8))), 256, 0, stream>>>(x, xbf);
  cvt_w_kernel<<<dim3(D_SZ / 32, D_SZ / 32, 3), 256, 0, stream>>>(Wq, Wk, Wv, Wtb);

  qkv_gemm_kernel<<<dim3(D_SZ / 64, (B_SZ * S_SZ) / 128, 3), 256, 0, stream>>>(
      xbf, Wtb, bq, bk, bv, Qb, Kb, Vt);

  attn_kernel<<<dim3(S_SZ / 16, H_SZ, B_SZ), 32, 0, stream>>>(Qb, Kb, Vt, out);
}
